// Rhn_53231824667307
// MI455X (gfx1250) — matrix operand order repaired and checked by reading, not compiled, not run on hardware
//
#include <hip/hip_runtime.h>

// ---------------- problem constants ----------------
#define Bq 64
#define Tq 256
#define Dq 1024
#define Hq 1024
#define Lq 5
#define BT (Bq * Tq)          // 16384
#define PERMAT (Dq * Hq)      // 1024*1024 elements per weight matrix

// ---------------- WMMA types ----------------
typedef __attribute__((ext_vector_type(16))) __bf16 v16bf;
typedef __attribute__((ext_vector_type(8)))  float  v8f;

union FragBF { uint4 q[2]; unsigned short u[16]; v16bf v; };

__device__ __forceinline__ unsigned short f2bf(float f) {
    union { float f; unsigned u; } x; x.f = f;
    unsigned r = x.u + 0x7FFFu + ((x.u >> 16) & 1u);
    return (unsigned short)(r >> 16);
}

__device__ __forceinline__ v8f wmma_bf16(v16bf a, v16bf b, v8f c) {
    return __builtin_amdgcn_wmma_f32_16x16x32_bf16(
        /*neg_a=*/false, a, /*neg_b=*/false, b,
        /*c_mod=*/(short)0, c, /*reuse_a=*/false, /*reuse_b=*/false);
}

__device__ __forceinline__ float fast_sigmoid(float x) {
    return __frcp_rn(1.0f + __expf(-x));
}

// ---------------- grid-wide barrier (persistent kernel) ----------------
__device__ __forceinline__ void grid_barrier(int* bar) {
    __syncthreads();
    if (threadIdx.x == 0) {
        __threadfence();
        volatile int* vbar = (volatile int*)bar;
        int gen = vbar[1];
        int prev = atomicAdd(bar, 1);
        if (prev == (int)gridDim.x - 1) {
            bar[0] = 0;
            __threadfence();
            atomicAdd(bar + 1, 1);
        } else {
            while (vbar[1] == gen) { __builtin_amdgcn_s_sleep(1); }
        }
        __threadfence();
    }
    __syncthreads();
}

// ---------------------------------------------------------------------------
// Kernel 1: convert fp32 weights -> bf16, pre-packed in WMMA B-fragment order.
// Fragment (kTile=32 K x 16 N): lane<16 -> K = kk*32 + {0..7, 16..23} (e ascending),
// lane>=16 -> K = kk*32 + {8..15, 24..31}; n = nTile*16 + (lane&15).
// Packed flat index within a matrix: r = ((nTile*32 + kk)*32 + lane)*16 + e.
// ---------------------------------------------------------------------------
__global__ void __launch_bounds__(256) pack_weights_kernel(
    const float* __restrict__ w_h, const float* __restrict__ w_t, const float* __restrict__ w_c,
    const float* __restrict__ r_hw, const float* __restrict__ r_tw, const float* __restrict__ r_cw,
    unsigned short* __restrict__ packedW, unsigned short* __restrict__ packedR)
{
    unsigned gid = blockIdx.x * 256u + threadIdx.x;   // 18M total
    if (gid >= 18u * PERMAT) return;
    int mat = (int)(gid >> 20);
    int r   = (int)(gid & (PERMAT - 1));
    int e    = r & 15;
    int lane = (r >> 4) & 31;
    int kk   = (r >> 9) & 31;
    int nT   = r >> 14;
    int k = kk * 32 + ((lane >> 4) << 3) + (e & 7) + ((e >> 3) << 4);
    int n = nT * 16 + (lane & 15);

    const float* src;
    unsigned short* dst;
    if (mat < 3) {
        src = (mat == 0) ? w_h : (mat == 1) ? w_t : w_c;
        dst = packedW + (size_t)mat * PERMAT;
    } else {
        int lm = mat - 3;             // lm = l*3 + j
        int j = lm % 3;
        int l = lm / 3;
        const float* rw = (j == 0) ? r_hw : (j == 1) ? r_tw : r_cw;
        src = rw + (size_t)l * PERMAT;
        dst = packedR + (size_t)lm * PERMAT;
    }
    dst[r] = f2bf(src[(size_t)k * Hq + n]);
}

// ---------------------------------------------------------------------------
// Kernel 2: init s state (parity 0) and grid-barrier counters
// ---------------------------------------------------------------------------
__global__ void __launch_bounds__(256) init_state_kernel(
    const float* __restrict__ s0, float* __restrict__ s32,
    unsigned short* __restrict__ s16, int* __restrict__ bar)
{
    int i = blockIdx.x * 256 + threadIdx.x;
    if (i < Bq * Hq) {
        float v = s0[i];
        s32[i] = v;
        s16[i] = f2bf(v);
    }
    if (i < 8) bar[i] = 0;
}

// ---------------------------------------------------------------------------
// Kernel 3: batched input projections  P[j] = X @ W_j  (fp32 in, bf16 WMMA,
// fp32 out).  X viewed as [BT, D] row-major (row = b*T + t).
// Each wave owns one (mTile,nTile) and computes all 3 gate projections,
// reusing the A fragment across the 3 WMMAs.
// ---------------------------------------------------------------------------
__global__ void __launch_bounds__(256) proj_kernel(
    const float* __restrict__ x, const unsigned short* __restrict__ packedW,
    float* __restrict__ P)
{
    const int nWavesBlk = blockDim.x >> 5;
    int wave  = blockIdx.x * nWavesBlk + (int)(threadIdx.x >> 5);
    int lane  = threadIdx.x & 31;
    int nWaves = gridDim.x * nWavesBlk;
    int halfSel = lane >> 4;
    const size_t PL = (size_t)BT * Hq;          // plane stride

    const int nTiles = (BT / 16) * (Hq / 16);   // 1024 * 64

    for (int tt = wave; tt < nTiles; tt += nWaves) {
        int mT = tt >> 6;
        int nT = tt & 63;
        const float* xr = x + (size_t)(mT * 16 + (lane & 15)) * Dq;

        v8f acc0 = {}; v8f acc1 = {}; v8f acc2 = {};
        for (int kk = 0; kk < 32; ++kk) {
            int k0 = kk * 32 + halfSel * 8;
            // A fragment: fp32 -> bf16 on the fly (contiguous per 8-K run)
            FragBF a;
            const float* p0 = xr + k0;
            const float* p1 = xr + k0 + 16;
#pragma unroll
            for (int e = 0; e < 8; ++e) a.u[e]     = f2bf(p0[e]);
#pragma unroll
            for (int e = 0; e < 8; ++e) a.u[8 + e] = f2bf(p1[e]);

            size_t fb = ((size_t)(nT * 32 + kk) << 9) + ((size_t)lane << 4);
            const uint4* b0 = (const uint4*)(packedW + 0 * (size_t)PERMAT + fb);
            const uint4* b1 = (const uint4*)(packedW + 1 * (size_t)PERMAT + fb);
            const uint4* b2 = (const uint4*)(packedW + 2 * (size_t)PERMAT + fb);
            FragBF fB0; fB0.q[0] = b0[0]; fB0.q[1] = b0[1];
            FragBF fB1; fB1.q[0] = b1[0]; fB1.q[1] = b1[1];
            FragBF fB2; fB2.q[0] = b2[0]; fB2.q[1] = b2[1];

            acc0 = wmma_bf16(a.v, fB0.v, acc0);
            acc1 = wmma_bf16(a.v, fB1.v, acc1);
            acc2 = wmma_bf16(a.v, fB2.v, acc2);
        }
        int col = nT * 16 + (lane & 15);
#pragma unroll
        for (int i = 0; i < 8; ++i) {
            int row = mT * 16 + halfSel * 8 + i;
            size_t o = (size_t)row * Hq + col;
            P[o]          = acc0[i];
            P[PL + o]     = acc1[i];
            P[2 * PL + o] = acc2[i];
        }
    }
}

// ---------------------------------------------------------------------------
// Kernel 4: persistent recurrent kernel. 32 WGs x 256 threads = 256 waves.
// wave w -> (mTile = w>>6 in [0,4), nTile = w&63 in [0,64)).
// Each wave computes the SAME 16x16 tile of Gh, Gt, Gc (A frag reused 3x),
// applies the highway nonlinearity, and writes s (ping-pong fp32 + bf16).
// One grid barrier per (t,l) iteration.
// ---------------------------------------------------------------------------
__global__ void __launch_bounds__(256, 1) rhn_kernel(
    const unsigned short* __restrict__ packedR,   // 15 matrices, fragment-packed
    const float* __restrict__ P,                  // 3 planes [BT, H]
    const float* __restrict__ r_hb, const float* __restrict__ r_tb,
    const float* __restrict__ r_cb,
    float* __restrict__ s32, unsigned short* __restrict__ s16,
    float* __restrict__ out, int* __restrict__ bar)
{
    int wave = blockIdx.x * (blockDim.x >> 5) + (int)(threadIdx.x >> 5); // 0..255
    int lane = threadIdx.x & 31;
    int mTile = wave >> 6;           // 0..3   (batch rows)
    int nTile = wave & 63;           // 0..63  (H columns)
    int halfSel = lane >> 4;
    int col  = nTile * 16 + (lane & 15);
    int aRow = mTile * 16 + (lane & 15);
    const size_t PL = (size_t)BT * Hq;
    const int SN = Bq * Hq;

    for (int t = 0; t < Tq; ++t) {
        for (int l = 0; l < Lq; ++l) {
            int iter = t * Lq + l;
            int pr = iter & 1;                      // read parity
            grid_barrier(bar);                      // prev writes visible

            const unsigned short* sIn16 = s16 + (size_t)pr * SN;
            const float*          sIn32 = s32 + (size_t)pr * SN;
            const unsigned short* Rbase = packedR + (size_t)l * 3 * PERMAT;

            v8f accH = {}; v8f accT = {}; v8f accC = {};
            for (int kk = 0; kk < 32; ++kk) {
                int k0 = kk * 32 + halfSel * 8;
                const unsigned short* ap = sIn16 + (size_t)aRow * Hq + k0;
                FragBF a;
                a.q[0] = *(const uint4*)(ap);
                a.q[1] = *(const uint4*)(ap + 16);

                size_t fb = ((size_t)(nTile * 32 + kk) << 9) + ((size_t)lane << 4);
                const uint4* bh = (const uint4*)(Rbase + 0 * (size_t)PERMAT + fb);
                const uint4* bt = (const uint4*)(Rbase + 1 * (size_t)PERMAT + fb);
                const uint4* bc = (const uint4*)(Rbase + 2 * (size_t)PERMAT + fb);
                FragBF fH; fH.q[0] = bh[0]; fH.q[1] = bh[1];
                FragBF fT; fT.q[0] = bt[0]; fT.q[1] = bt[1];
                FragBF fC; fC.q[0] = bc[0]; fC.q[1] = bc[1];

                if (kk + 1 < 32) {
                    // pull next K-slice of packed weights toward the WGP
                    __builtin_prefetch((const void*)(Rbase + fb + 512), 0, 0);
                }

                accH = wmma_bf16(a.v, fH.v, accH);
                accT = wmma_bf16(a.v, fT.v, accT);
                accC = wmma_bf16(a.v, fC.v, accC);
            }

            // ----- elementwise highway update (writes go to parity pr^1) ----
            float bhv = r_hb[l * Hq + col];
            float btv = r_tb[l * Hq + col];
            float bcv = r_cb[l * Hq + col];
            float*          sOut32 = s32 + (size_t)(pr ^ 1) * SN;
            unsigned short* sOut16 = s16 + (size_t)(pr ^ 1) * SN;

#pragma unroll
            for (int i = 0; i < 8; ++i) {
                int b   = mTile * 16 + halfSel * 8 + i;    // batch index
                int idx = b * Hq + col;
                float gh = accH[i] + bhv;
                float gt = accT[i] + btv;
                float gc = accC[i] + bcv;
                if (l == 0) {
                    size_t po = (size_t)(b * Tq + t) * Hq + col;
                    gh += P[po];
                    gt += P[PL + po];
                    gc += P[2 * PL + po];
                }
                float sOld = sIn32[idx];
                float hv = tanhf(gh);
                float tv = fast_sigmoid(gt);
                float cv = fast_sigmoid(gc);
                float sN = hv * tv + sOld * cv;
                sOut32[idx] = sN;
                sOut16[idx] = f2bf(sN);
                if (l == Lq - 1) {
                    out[(size_t)(b * Tq + t) * Hq + col] = sN;       // outs[B,T,H]
                    if (t == Tq - 1) out[(size_t)Bq * Tq * Hq + idx] = sN; // s_final
                }
            }
        }
    }
}

// ---------------------------------------------------------------------------
// Host launcher
// ---------------------------------------------------------------------------
extern "C" void kernel_launch(void* const* d_in, const int* in_sizes, int n_in,
                              void* d_out, int out_size, void* d_ws, size_t ws_size,
                              hipStream_t stream) {
    (void)in_sizes; (void)n_in; (void)out_size; (void)ws_size;
    const float* x    = (const float*)d_in[0];
    const float* s0   = (const float*)d_in[1];
    const float* w_h  = (const float*)d_in[2];
    const float* w_t  = (const float*)d_in[3];
    const float* w_c  = (const float*)d_in[4];
    const float* r_hw = (const float*)d_in[5];
    const float* r_tw = (const float*)d_in[6];
    const float* r_cw = (const float*)d_in[7];
    const float* r_hb = (const float*)d_in[8];
    const float* r_tb = (const float*)d_in[9];
    const float* r_cb = (const float*)d_in[10];
    float* out = (float*)d_out;

    // workspace layout (bytes, all 16B-aligned)
    char* ws = (char*)d_ws;
    const size_t OFF_PW  = 0;                                  // 3  * 1M bf16 =   6 MB
    const size_t OFF_PR  = OFF_PW + 3ull  * PERMAT * 2;        // 15 * 1M bf16 =  30 MB
    const size_t OFF_P   = OFF_PR + 15ull * PERMAT * 2;        // 3*BT*H fp32  = 192 MB
    const size_t OFF_S32 = OFF_P  + 3ull * BT * Hq * 4;        // 2*B*H fp32
    const size_t OFF_S16 = OFF_S32 + 2ull * Bq * Hq * 4;       // 2*B*H bf16
    const size_t OFF_BAR = OFF_S16 + 2ull * Bq * Hq * 2;       // barrier ints

    unsigned short* packedW = (unsigned short*)(ws + OFF_PW);
    unsigned short* packedR = (unsigned short*)(ws + OFF_PR);
    float*          Pbuf    = (float*)(ws + OFF_P);
    float*          s32     = (float*)(ws + OFF_S32);
    unsigned short* s16     = (unsigned short*)(ws + OFF_S16);
    int*            bar     = (int*)(ws + OFF_BAR);

    // 1) pack weights to bf16 WMMA-fragment layout (18 * 1M elements)
    pack_weights_kernel<<<(18u * PERMAT) / 256u, 256, 0, stream>>>(
        w_h, w_t, w_c, r_hw, r_tw, r_cw, packedW, packedR);

    // 2) init state + barrier
    init_state_kernel<<<(Bq * Hq + 255) / 256, 256, 0, stream>>>(s0, s32, s16, bar);

    // 3) batched input projections
    proj_kernel<<<256, 256, 0, stream>>>(x, packedW, Pbuf);

    // 4) persistent recurrent kernel (32 WGs -> 256 waves, 1 tile-triple each)
    rhn_kernel<<<32, 256, 0, stream>>>(packedR, Pbuf, r_hb, r_tb, r_cb,
                                       s32, s16, out, bar);
}
